// GroupedKAAttention_87436944212733
// MI455X (gfx1250) — compile-verified, matrix operand order checked by reading, not run on hardware
//
#include <hip/hip_runtime.h>

// ---------------------------------------------------------------------------
// GroupedKAAttention for MI455X (gfx1250).
//
// Phase 0 (prep, bandwidth-bound):
//   - gather+convert q/k slices -> bf16, K padded 1176->1184 (zero fill)
//   - convert+transpose all weights -> bf16 column-major [n][k]
// Phase 1 (compute, WMMA):
//   - generic bf16 GEMM, 128x128 macro-tile, BK=32, 8 waves
//   - tiles staged global->LDS by the Tensor Data Mover (tensor_load_to_lds,
//     TENSORcnt, LDS padding via the TDM pad feature), double-buffered so the
//     DMA of tile t+1 overlaps v_wmma on tile t
//   - epilogue fuses bias(+relu) and writes bf16 intermediates / fp32 finals
// Phase 2: attn[b] = dot(qo,ko); softmax over a singleton axis == 1 exactly.
// ---------------------------------------------------------------------------

#ifndef USE_TDM
#define USE_TDM 1
#endif

namespace {
constexpr int kB   = 512;
constexpr int kG   = 4;
constexpr int kIN  = 1176;
constexpr int kKp1 = 1184;        // kIN padded to a multiple of BK
constexpr int kH   = 1024;
constexpr int kOUT = 512;
constexpr int kGIN = kOUT * kG;   // 2048

constexpr int BM   = 128;
constexpr int BN   = 128;
constexpr int BK   = 32;
constexpr int TPB  = 256;         // 8 waves (wave32)
constexpr int LSTR = BK + 8;      // LDS row stride in halves (80 B) == TDM pad
}

typedef __attribute__((ext_vector_type(8)))  float        v8f;
typedef __attribute__((ext_vector_type(8)))  __bf16       v8bf;
typedef __attribute__((ext_vector_type(16))) __bf16       v16bf;
typedef __attribute__((ext_vector_type(4)))  unsigned int u32x4;
typedef __attribute__((ext_vector_type(4)))  int          i32x4;
typedef __attribute__((ext_vector_type(8)))  int          i32x8;

static __device__ __forceinline__ unsigned short f2bf(float f) {
  unsigned int u = __float_as_uint(f);
  u += 0x7FFFu + ((u >> 16) & 1u);   // round-to-nearest-even
  return (unsigned short)(u >> 16);
}

// Generic LDS address -> 32-bit LDS byte offset (flat LDS aperture keeps the
// offset in addr[31:0], per CDNA5 ISA 10.2).
static __device__ __forceinline__ unsigned lds_off_of(const void* p) {
  return (unsigned)(unsigned long long)p;
}

#if USE_TDM
// One 2D TDM transfer: tile_k x tile_rows bf16 tile, global row stride
// row_stride elements, into LDS at lds_off with 16B padding every 64B row
// (-> effective LDS row stride of 40 halves).
static __device__ __forceinline__ void tdm_load_2d(unsigned lds_off,
                                                   const unsigned short* gptr,
                                                   int tile_k, int tile_rows,
                                                   long long row_stride,
                                                   int tensor_k) {
  const unsigned long long ga = (unsigned long long)gptr;
  u32x4 g0;
  g0[0] = 1u;                                   // count=1 (valid), user mode
  g0[1] = lds_off;                              // LDS byte address
  g0[2] = (unsigned)(ga & 0xFFFFFFFFull);       // global addr [31:0]
  g0[3] = (unsigned)((ga >> 32) & 0x01FFFFFFull) | (2u << 30);  // [56:32]|type=2

  const unsigned long long s0 = (unsigned long long)row_stride;  // dim0 stride
  const unsigned td0 = (unsigned)tensor_k;      // tensor dim0 (elements)
  const unsigned td1 = (unsigned)tile_rows;     // tensor dim1 (rows)
  i32x8 g1;
  // data_size=1 (2B), pad_enable, pad_interval=3 (16 DW), pad_amount=3 (4 DW)
  g1[0] = (int)((1u << 16) | (1u << 20) | (3u << 22) | (3u << 25));
  g1[1] = (int)((td0 & 0xFFFFu) << 16);                    // td0[15:0]
  g1[2] = (int)((td0 >> 16) | ((td1 & 0xFFFFu) << 16));    // td0[31:16], td1 lo
  g1[3] = (int)((td1 >> 16) | ((unsigned)tile_k << 16));   // td1 hi, tile_dim0
  g1[4] = (int)(unsigned)tile_rows;                        // tile_dim1 (dim2=0)
  g1[5] = (int)(unsigned)(s0 & 0xFFFFFFFFull);             // dim0 stride lo32
  g1[6] = (int)(unsigned)((s0 >> 32) & 0xFFFFu);           // dim0 stride hi16
  g1[7] = 0;

  const i32x4 z4 = {0, 0, 0, 0};
#if defined(__clang_major__) && __clang_major__ >= 23
  const i32x8 z8 = {0, 0, 0, 0, 0, 0, 0, 0};
  __builtin_amdgcn_tensor_load_to_lds(g0, g1, z4, z4, z8, 0);
#else
  __builtin_amdgcn_tensor_load_to_lds(g0, g1, z4, z4, 0);
#endif
}
#endif  // USE_TDM

// ---------------------------------------------------------------------------
// Prep kernels
// ---------------------------------------------------------------------------

// out[(g*B+b)][0..Kp1) = bf16(q[b, g*IN .. g*IN+IN)), zero-padded to Kp1.
__global__ __launch_bounds__(TPB) void pack_activation(
    const float* __restrict__ src, unsigned short* __restrict__ dst) {
  const int row = blockIdx.x;          // g*B + b
  const int g = row >> 9;              // /kB
  const int b = row & (kB - 1);
  const float* s = src + (long long)b * (128LL * kIN) + (long long)g * kIN;
  unsigned short* d = dst + (long long)row * kKp1;
  for (int i = threadIdx.x; i < kKp1; i += TPB)
    d[i] = (i < kIN) ? f2bf(s[i]) : (unsigned short)0;
}

// dst[n][k] = bf16(src[k][n]); src is (K_,N_) fp32, dst is (N_,Kp_) bf16,
// zero pad for k in [K_,Kp_). N_ must be a multiple of 32.
__global__ __launch_bounds__(TPB) void transpose_convert(
    const float* __restrict__ src0, unsigned short* __restrict__ dst0,
    int K_, int N_, int Kp_, long long srcZ, long long dstZ) {
  __shared__ float t[32][33];
  const float* src = src0 + (long long)blockIdx.z * srcZ;
  unsigned short* dst = dst0 + (long long)blockIdx.z * dstZ;
  const int tx = threadIdx.x & 31;
  const int ty = threadIdx.x >> 5;       // 0..7
  const int k0 = blockIdx.x * 32;
  const int n0 = blockIdx.y * 32;
  #pragma unroll
  for (int j = 0; j < 4; ++j) {
    const int kk = k0 + ty + j * 8;
    t[ty + j * 8][tx] = (kk < K_) ? src[(long long)kk * N_ + n0 + tx] : 0.f;
  }
  __syncthreads();
  #pragma unroll
  for (int j = 0; j < 4; ++j) {
    const int n = n0 + ty + j * 8;
    dst[(long long)n * Kp_ + k0 + tx] = f2bf(t[tx][ty + j * 8]);
  }
}

// ---------------------------------------------------------------------------
// bf16 WMMA GEMM, TDM-staged, double-buffered LDS
// ---------------------------------------------------------------------------

struct GemmP {
  const unsigned short* A;   long long lda, Az;   // bf16 row-major [m][k]
  const unsigned short* Bt;  long long ldb, Bz;   // bf16 col-major  [n][k]
  const float* bias;         long long biasz;     // fp32 bias[n]
  void* C;                   long long ldc, Cz;   // C[m*ldc + n*colStride]
  int colStride, M, N, K, relu, outBf16;
};

__global__ __launch_bounds__(TPB) void gemm_bf16_wmma(GemmP p) {
  __shared__ unsigned short As[2][BM][LSTR];
  __shared__ unsigned short Bs[2][BN][LSTR];

  const int z = blockIdx.z;
  const unsigned short* Ag = p.A  + (long long)z * p.Az;
  const unsigned short* Bg = p.Bt + (long long)z * p.Bz;
  const float* bsp = p.bias ? (p.bias + (long long)z * p.biasz) : nullptr;

  const int m0 = blockIdx.y * BM;
  const int n0 = blockIdx.x * BN;

  const int tid   = threadIdx.x;
  const int lane  = tid & 31;
  const int wave  = tid >> 5;
  const int lr    = lane & 15;
  const int hf    = lane >> 4;
  const int waveM = (wave & 3) * 32;
  const int waveN = (wave >> 2) * 64;

  const v8f zeroV = {0.f, 0.f, 0.f, 0.f, 0.f, 0.f, 0.f, 0.f};
  v8f acc[2][4];
  #pragma unroll
  for (int mi = 0; mi < 2; ++mi)
    #pragma unroll
    for (int ni = 0; ni < 4; ++ni) acc[mi][ni] = zeroV;

  const int nt = p.K / BK;               // K is always a multiple of BK

#if USE_TDM
  if (wave == 0) {
    tdm_load_2d(lds_off_of(&As[0][0][0]), Ag + (long long)m0 * p.lda,
                BK, BM, p.lda, p.K);
    tdm_load_2d(lds_off_of(&Bs[0][0][0]), Bg + (long long)n0 * p.ldb,
                BK, BN, p.ldb, p.K);
  }
#endif

  for (int t = 0; t < nt; ++t) {
    const int cur = t & 1;
#if USE_TDM
    if (wave == 0) {
      if (t + 1 < nt) {                  // prefetch next tile via TDM
        const int kb = (t + 1) * BK;
        tdm_load_2d(lds_off_of(&As[cur ^ 1][0][0]),
                    Ag + (long long)m0 * p.lda + kb, BK, BM, p.lda, p.K);
        tdm_load_2d(lds_off_of(&Bs[cur ^ 1][0][0]),
                    Bg + (long long)n0 * p.ldb + kb, BK, BN, p.ldb, p.K);
        __builtin_amdgcn_s_wait_tensorcnt(2);   // tile t complete
      } else {
        __builtin_amdgcn_s_wait_tensorcnt(0);
      }
    }
#else
    {   // manual bf16 tile copy: 32 B per thread per operand, no guards
      const int kb = t * BK;
      const int r = tid >> 1, h = (tid & 1) * 16;
      const unsigned short* sa = Ag + (long long)(m0 + r) * p.lda + kb + h;
      const uint4 a0 = *reinterpret_cast<const uint4*>(sa);
      const uint4 a1 = *reinterpret_cast<const uint4*>(sa + 8);
      *reinterpret_cast<uint4*>(&As[cur][r][h]) = a0;
      *reinterpret_cast<uint4*>(&As[cur][r][h + 8]) = a1;
      const unsigned short* sb = Bg + (long long)(n0 + r) * p.ldb + kb + h;
      const uint4 b0 = *reinterpret_cast<const uint4*>(sb);
      const uint4 b1 = *reinterpret_cast<const uint4*>(sb + 8);
      *reinterpret_cast<uint4*>(&Bs[cur][r][h]) = b0;
      *reinterpret_cast<uint4*>(&Bs[cur][r][h + 8]) = b1;
    }
#endif
    __syncthreads();

    // Fragments per CDNA5 16-bit layouts: lanes 0-15 hold K 0..7 & 16..23,
    // lanes 16-31 hold K 8..15 & 24..31; M (resp. N) = lane%16.
    v16bf afrag[2];
    #pragma unroll
    for (int mi = 0; mi < 2; ++mi) {
      const unsigned short* ap = &As[cur][waveM + mi * 16 + lr][0];
      const v8bf lo = *reinterpret_cast<const v8bf*>(ap + hf * 8);
      const v8bf hi = *reinterpret_cast<const v8bf*>(ap + 16 + hf * 8);
      v16bf a;
      #pragma unroll
      for (int e = 0; e < 8; ++e) { a[e] = lo[e]; a[8 + e] = hi[e]; }
      afrag[mi] = a;
    }
    #pragma unroll
    for (int ni = 0; ni < 4; ++ni) {
      const unsigned short* bp = &Bs[cur][waveN + ni * 16 + lr][0];
      const v8bf lo = *reinterpret_cast<const v8bf*>(bp + hf * 8);
      const v8bf hi = *reinterpret_cast<const v8bf*>(bp + 16 + hf * 8);
      v16bf bfrag;
      #pragma unroll
      for (int e = 0; e < 8; ++e) { bfrag[e] = lo[e]; bfrag[8 + e] = hi[e]; }
      #pragma unroll
      for (int mi = 0; mi < 2; ++mi) {
        acc[mi][ni] = __builtin_amdgcn_wmma_f32_16x16x32_bf16(
            false, afrag[mi], false, bfrag, (short)0, acc[mi][ni],
            false, false);
      }
    }
    __syncthreads();   // protect buffer 'cur' before TDM t+2 overwrites it
  }

  // Epilogue: bias (+relu); bf16 intermediates or fp32 finals; optional
  // group-interleaved column stride.
  #pragma unroll
  for (int mi = 0; mi < 2; ++mi) {
    #pragma unroll
    for (int ni = 0; ni < 4; ++ni) {
      const int gn = n0 + waveN + ni * 16 + lr;
      const float bv = bsp ? bsp[gn] : 0.f;
      #pragma unroll
      for (int r = 0; r < 8; ++r) {
        const int gm = m0 + waveM + mi * 16 + hf * 8 + r;
        float v = acc[mi][ni][r] + bv;
        if (p.relu) v = v > 0.f ? v : 0.f;
        const long long idx =
            (long long)gm * p.ldc + (long long)gn * p.colStride;
        if (p.outBf16)
          ((unsigned short*)p.C + (long long)z * p.Cz)[idx] = f2bf(v);
        else
          ((float*)p.C + (long long)z * p.Cz)[idx] = v;
      }
    }
  }
}

// attn[b] = dot(qo[b], ko[b]); softmax over the singleton last axis == 1.
__global__ __launch_bounds__(TPB) void finalize_kernel(
    const float* __restrict__ Qo, const float* __restrict__ Ko,
    float* __restrict__ attn, float* __restrict__ out) {
  __shared__ float red[TPB];
  const int b = blockIdx.x;
  float s = 0.f;
  for (int o = threadIdx.x; o < kOUT; o += blockDim.x)
    s += Qo[(long long)b * kOUT + o] * Ko[(long long)b * kOUT + o];
  red[threadIdx.x] = s;
  __syncthreads();
  for (int st = TPB / 2; st > 0; st >>= 1) {
    if ((int)threadIdx.x < st) red[threadIdx.x] += red[threadIdx.x + st];
    __syncthreads();
  }
  if (threadIdx.x == 0) {
    attn[b] = red[0];
    out[b] = 1.0f;
  }
}

// ---------------------------------------------------------------------------
// Host-side orchestration
// ---------------------------------------------------------------------------

extern "C" void kernel_launch(void* const* d_in, const int* in_sizes, int n_in,
                              void* d_out, int out_size, void* d_ws, size_t ws_size,
                              hipStream_t stream) {
  (void)in_sizes; (void)n_in; (void)out_size; (void)ws_size;
  typedef unsigned short us;

  const float* q   = (const float*)d_in[0];
  const float* k   = (const float*)d_in[1];
  const float* Wq1 = (const float*)d_in[2];
  const float* bq1 = (const float*)d_in[3];
  const float* Wq2 = (const float*)d_in[4];
  const float* bq2 = (const float*)d_in[5];
  const float* Wk1 = (const float*)d_in[6];
  const float* bk1 = (const float*)d_in[7];
  const float* Wk2 = (const float*)d_in[8];
  const float* bk2 = (const float*)d_in[9];
  const float* Wg1 = (const float*)d_in[10];
  const float* bg1 = (const float*)d_in[11];
  const float* Wg2 = (const float*)d_in[12];
  const float* bg2 = (const float*)d_in[13];

  us* w = (us*)d_ws;
  size_t o = 0;
  us* XAq  = w + o; o += (size_t)kG * kB * kKp1;      // bf16 activations
  us* XAk  = w + o; o += (size_t)kG * kB * kKp1;
  us* W1tq = w + o; o += (size_t)kG * kH * kKp1;      // bf16 [n][k] weights
  us* W1tk = w + o; o += (size_t)kG * kH * kKp1;
  us* W2tq = w + o; o += (size_t)kG * kOUT * kH;
  us* W2tk = w + o; o += (size_t)kG * kOUT * kH;
  us* Wg1t = w + o; o += (size_t)kH * kGIN;
  us* Wg2t = w + o; o += (size_t)kOUT * kH;
  us* H1q  = w + o; o += (size_t)kB * kG * kH;        // bf16 intermediates
  us* H1k  = w + o; o += (size_t)kB * kG * kH;
  us* Fq   = w + o; o += (size_t)kB * kGIN;           // Fq,Fk contiguous
  us* Fk   = w + o; o += (size_t)kB * kGIN;
  us* Hgq  = w + o; o += (size_t)kB * kH;             // Hgq,Hgk contiguous
  us* Hgk  = w + o; o += (size_t)kB * kH;
  o = (o + 7) & ~(size_t)7;                            // 16B align for floats
  float* Qo   = (float*)(w + o); o += (size_t)2 * kB * kOUT;  // Qo,Ko (fp32)
  float* Ko   = Qo + (size_t)kB * kOUT;
  float* attn = Qo + (size_t)2 * kB * kOUT;
  (void)Fk; (void)Hgk;

  const dim3 blk(TPB);

  // ---- phase 0: conversions ----
  pack_activation<<<dim3(kG * kB), blk, 0, stream>>>(q, XAq);
  pack_activation<<<dim3(kG * kB), blk, 0, stream>>>(k, XAk);
  transpose_convert<<<dim3(kKp1 / 32, kH / 32, kG), blk, 0, stream>>>(
      Wq1, W1tq, kIN, kH, kKp1, (long long)kIN * kH, (long long)kH * kKp1);
  transpose_convert<<<dim3(kKp1 / 32, kH / 32, kG), blk, 0, stream>>>(
      Wk1, W1tk, kIN, kH, kKp1, (long long)kIN * kH, (long long)kH * kKp1);
  transpose_convert<<<dim3(kH / 32, kOUT / 32, kG), blk, 0, stream>>>(
      Wq2, W2tq, kH, kOUT, kH, (long long)kH * kOUT, (long long)kOUT * kH);
  transpose_convert<<<dim3(kH / 32, kOUT / 32, kG), blk, 0, stream>>>(
      Wk2, W2tk, kH, kOUT, kH, (long long)kH * kOUT, (long long)kOUT * kH);
  transpose_convert<<<dim3(kGIN / 32, kH / 32, 1), blk, 0, stream>>>(
      Wg1, Wg1t, kGIN, kH, kGIN, 0, 0);
  transpose_convert<<<dim3(kH / 32, kOUT / 32, 1), blk, 0, stream>>>(
      Wg2, Wg2t, kH, kOUT, kH, 0, 0);

  // ---- phase 1: GEMMs ----
  GemmP p1;   // grouped layer1 (z = group): (512 x 1184) x (1184 x 1024)
  p1.A = XAq;  p1.lda = kKp1; p1.Az = (long long)kB * kKp1;
  p1.Bt = W1tq; p1.ldb = kKp1; p1.Bz = (long long)kH * kKp1;
  p1.bias = bq1; p1.biasz = kH;
  p1.C = H1q; p1.ldc = (long long)kG * kH; p1.Cz = kH;
  p1.colStride = 1; p1.M = kB; p1.N = kH; p1.K = kKp1; p1.relu = 1;
  p1.outBf16 = 1;
  gemm_bf16_wmma<<<dim3(kH / BN, kB / BM, kG), blk, 0, stream>>>(p1);
  GemmP p1k = p1; p1k.A = XAk; p1k.Bt = W1tk; p1k.bias = bk1; p1k.C = H1k;
  gemm_bf16_wmma<<<dim3(kH / BN, kB / BM, kG), blk, 0, stream>>>(p1k);

  GemmP p2;   // grouped layer2 (z = group) -> interleaved f[b, o*G+g]
  p2.A = H1q; p2.lda = (long long)kG * kH; p2.Az = kH;
  p2.Bt = W2tq; p2.ldb = kH; p2.Bz = (long long)kOUT * kH;
  p2.bias = bq2; p2.biasz = kOUT;
  p2.C = Fq; p2.ldc = kGIN; p2.Cz = 1;
  p2.colStride = kG; p2.M = kB; p2.N = kOUT; p2.K = kH; p2.relu = 0;
  p2.outBf16 = 1;
  gemm_bf16_wmma<<<dim3(kOUT / BN, kB / BM, kG), blk, 0, stream>>>(p2);
  GemmP p2k = p2; p2k.A = H1k; p2k.Bt = W2tk; p2k.bias = bk2; p2k.C = Fk;
  gemm_bf16_wmma<<<dim3(kOUT / BN, kB / BM, kG), blk, 0, stream>>>(p2k);

  GemmP pg1;  // global layer1 (z = 0:q, 1:k)
  pg1.A = Fq; pg1.lda = kGIN; pg1.Az = (long long)kB * kGIN;
  pg1.Bt = Wg1t; pg1.ldb = kGIN; pg1.Bz = 0;
  pg1.bias = bg1; pg1.biasz = 0;
  pg1.C = Hgq; pg1.ldc = kH; pg1.Cz = (long long)kB * kH;
  pg1.colStride = 1; pg1.M = kB; pg1.N = kH; pg1.K = kGIN; pg1.relu = 1;
  pg1.outBf16 = 1;
  gemm_bf16_wmma<<<dim3(kH / BN, kB / BM, 2), blk, 0, stream>>>(pg1);

  GemmP pg2;  // global layer2 (z = 0:q, 1:k) -> fp32 Qo/Ko
  pg2.A = Hgq; pg2.lda = kH; pg2.Az = (long long)kB * kH;
  pg2.Bt = Wg2t; pg2.ldb = kH; pg2.Bz = 0;
  pg2.bias = bg2; pg2.biasz = 0;
  pg2.C = Qo; pg2.ldc = kOUT; pg2.Cz = (long long)kB * kOUT;
  pg2.colStride = 1; pg2.M = kB; pg2.N = kOUT; pg2.K = kH; pg2.relu = 0;
  pg2.outBf16 = 0;
  gemm_bf16_wmma<<<dim3(kOUT / BN, kB / BM, 2), blk, 0, stream>>>(pg2);

  // ---- phase 2 ----
  finalize_kernel<<<dim3(kB), blk, 0, stream>>>(Qo, Ko, attn, (float*)d_out);
}
